// GNNCriticVariable_12094627905888
// MI455X (gfx1250) — matrix operand order repaired and checked by reading, not compiled, not run on hardware
//
#include <hip/hip_runtime.h>
#include <hip/hip_bf16.h>
#include <math.h>

typedef __attribute__((ext_vector_type(16))) _Float16 v16h;
typedef __attribute__((ext_vector_type(8)))  float    v8f;
typedef __attribute__((ext_vector_type(4)))  float    f32x4;
typedef int v4i __attribute__((vector_size(16)));

#define BB   64
#define NN   512
#define FF   128
#define HH   512
#define KK   16
#define MTOT (BB * NN)   // 32768 rows total

#if defined(__has_builtin)
# if __has_builtin(__builtin_amdgcn_global_load_async_to_lds_b128)
#  define HAVE_ASYNC_LDS 1
# endif
#endif

// ---- helpers -------------------------------------------------------------

// Assemble a v16h fragment from two 16-byte chunks (global or LDS).
__device__ inline v16h ld16(const _Float16* p0, const _Float16* p1) {
  v16h r;
  ((f32x4*)&r)[0] = *(const f32x4*)p0;
  ((f32x4*)&r)[1] = *(const f32x4*)p1;
  return r;
}

__device__ inline v8f wmma_f16(v16h a, v16h b, v8f c) {
  // (neg_a, A, neg_b, B, c_mod, C, reuse_a, reuse_b)
  return __builtin_amdgcn_wmma_f32_16x16x32_f16(false, a, false, b, (short)0, c,
                                                false, false);
}

#if defined(HAVE_ASYNC_LDS)
typedef __attribute__((address_space(1))) v4i* gvec_p;
typedef __attribute__((address_space(3))) v4i* lvec_p;

__device__ inline void async_copy16(const _Float16* gp, _Float16* lp) {
  __attribute__((address_space(1))) void* g1 =
      (__attribute__((address_space(1))) void*)(void*)gp;
  __attribute__((address_space(3))) void* l3 =
      (__attribute__((address_space(3))) void*)(void*)lp;
  __builtin_amdgcn_global_load_async_to_lds_b128((gvec_p)g1, (lvec_p)l3, 0, 0);
}
#endif

// ---- 1) f32 -> f16 conversions ------------------------------------------

__global__ void k_cvt16(const float* __restrict__ in, _Float16* __restrict__ out, int n) {
  int i = blockIdx.x * blockDim.x + threadIdx.x;
  if (i < n) out[i] = (_Float16)in[i];
}

// W[k][n] (f32, row-major KxH) -> WT[n][k] (f16, transposed): makes the GEMM
// weight staging 16B-contiguous on both global and LDS sides.
__global__ void k_cvtW_T(const float* __restrict__ in, _Float16* __restrict__ out,
                         int Kdim, int total) {
  int i = blockIdx.x * blockDim.x + threadIdx.x;
  if (i >= total) return;
  int k = i / HH, n = i - k * HH;
  out[(size_t)n * Kdim + k] = (_Float16)in[i];
}

// ---- 2) kNN (top K+1 smallest d2, drop first) ---------------------------

__global__ void k_topk(const float* __restrict__ obs, int* __restrict__ nbrs) {
  __shared__ float sx[NN], sy[NN];
  const int b = blockIdx.x, n = threadIdx.x;
  const float* row = obs + ((size_t)b * NN + n) * FF;
  const float px = row[0], py = row[1];
  sx[n] = px; sy[n] = py;
  __syncthreads();

  float bd[KK + 1]; int bi[KK + 1];
  #pragma unroll
  for (int i = 0; i <= KK; ++i) { bd[i] = 3.0e38f; bi[i] = 0; }

  for (int j = 0; j < NN; ++j) {
    const float dx = px - sx[j], dy = py - sy[j];
    const float d2 = dx * dx + dy * dy;
    if (d2 < bd[KK]) {                 // strict < : stable ties -> lower index first
      int p = KK;
      while (p > 0 && d2 < bd[p - 1]) {
        bd[p] = bd[p - 1]; bi[p] = bi[p - 1]; --p;
      }
      bd[p] = d2; bi[p] = j;
    }
  }
  int* onb = nbrs + ((size_t)b * NN + n) * KK;
  for (int k = 0; k < KK; ++k) onb[k] = bi[k + 1];   // drop self / nearest
}

// ---- 3) WMMA GEMM: out[M,512] = A[M,KDIM] * W[KDIM,512], f16 in / f32 out
// WT is the transposed f16 weight [512][KDIM].

template <int KDIM>
__global__ __launch_bounds__(256) void k_gemm(const _Float16* __restrict__ A,
                                              const _Float16* __restrict__ WT,
                                              float* __restrict__ outM,
                                              float* __restrict__ outA) {
  constexpr int P = KDIM + 8;                 // padded pitch (f16), keeps 16B align
  __shared__ _Float16 ldsW[32 * P];           // W strip transposed: [col][k]
  const int n0 = blockIdx.y * 32;

  // stage 32-column weight strip: 16-byte chunks, contiguous on both sides
  constexpr int CH = KDIM / 8;                // 16B chunks per column
  for (int idx = threadIdx.x; idx < 32 * CH; idx += 256) {
    const int c  = idx / CH;
    const int kb = (idx - c * CH) * 8;
    const _Float16* gp = WT + (size_t)(n0 + c) * KDIM + kb;
    _Float16* lp = &ldsW[c * P + kb];
#if defined(HAVE_ASYNC_LDS)
    async_copy16(gp, lp);
#else
    *(f32x4*)lp = *(const f32x4*)gp;
#endif
  }
#if defined(HAVE_ASYNC_LDS)
# if __has_builtin(__builtin_amdgcn_s_wait_asynccnt)
  __builtin_amdgcn_s_wait_asynccnt(0);
# else
  asm volatile("s_wait_asynccnt 0x0" ::: "memory");
# endif
#endif
  __syncthreads();

  const int wave = threadIdx.x >> 5;          // 8 waves, 32 rows each
  const int lane = threadIdx.x & 31;
  const int r  = lane & 15;
  const int hb = lane >> 4;
  const int mBase = blockIdx.x * 256 + wave * 32;
  const _Float16* Arow0 = A + (size_t)(mBase + r) * KDIM;
  const _Float16* Arow1 = Arow0 + (size_t)16 * KDIM;

  v8f acc00 = {}, acc01 = {}, acc10 = {}, acc11 = {};
  for (int k0 = 0; k0 < KDIM; k0 += 32) {
    // A 16x32 f16 fragment: elems 0-15 -> K=k0+hb*16.. (contiguous)
    v16h a0 = ld16(Arow0 + k0 + hb * 16, Arow0 + k0 + hb * 16 + 8);
    v16h a1 = ld16(Arow1 + k0 + hb * 16, Arow1 + k0 + hb * 16 + 8);
    // B 32x16 fragment: lane half selects K group of 16, contiguous in LDS
    const _Float16* l0 = &ldsW[(r)      * P + k0 + hb * 16];
    const _Float16* l1 = &ldsW[(16 + r) * P + k0 + hb * 16];
    v16h b0 = ld16(l0, l0 + 8);
    v16h b1 = ld16(l1, l1 + 8);
    acc00 = wmma_f16(a0, b0, acc00);
    acc01 = wmma_f16(a0, b1, acc01);
    acc10 = wmma_f16(a1, b0, acc10);
    acc11 = wmma_f16(a1, b1, acc11);
  }

  // C/D layout: VGPR v -> row (hb*8 + v), col = lane%16
  #pragma unroll
  for (int v = 0; v < 8; ++v) {
    const int row0 = mBase + hb * 8 + v;
    const size_t o00 = (size_t)row0 * HH + n0 + r;
    const size_t o10 = o00 + (size_t)16 * HH;
    const float c00 = acc00[v], c01 = acc01[v], c10 = acc10[v], c11 = acc11[v];
    outM[o00]      = c00; outA[o00]      = c00;  // a starts as m; scatter adds on outA
    outM[o00 + 16] = c01; outA[o00 + 16] = c01;
    outM[o10]      = c10; outA[o10]      = c10;
    outM[o10 + 16] = c11; outA[o10 + 16] = c11;
  }
}

// ---- 4) scatter: a[b, nbr[n,k], :] += m[b, n, :] -------------------------

__global__ void k_scatter(const float* __restrict__ m, const int* __restrict__ nbrs,
                          float* __restrict__ a) {
  const int node = blockIdx.x;            // 0..MTOT-1
  const int b = node >> 9;                // node / NN
  const float* mrow = m + (size_t)node * HH;
  const int d = threadIdx.x * 2;
  const float m0 = mrow[d], m1 = mrow[d + 1];

  __shared__ int snb[KK];
  if (threadIdx.x < KK) snb[threadIdx.x] = nbrs[(size_t)node * KK + threadIdx.x];
  __syncthreads();

  #pragma unroll
  for (int k = 0; k < KK; ++k) {
    float* arow = a + ((size_t)(b * NN + snb[k])) * HH + d;
    atomicAdd(arow,     m0);
    atomicAdd(arow + 1, m1);
  }
}

// ---- 5) activation: tanh(a + bias) -> f16 (and optional f32) -------------

__global__ void k_act(const float* __restrict__ a, const float* __restrict__ bias,
                      _Float16* __restrict__ o16, float* __restrict__ o32, int total) {
  const int i = blockIdx.x * blockDim.x + threadIdx.x;
  if (i >= total) return;
  const float t = tanhf(a[i] + bias[i & (HH - 1)]);
  o16[i] = (_Float16)t;
  if (o32) o32[i] = t;
}

// ---- 6) masked GEMV: out = mask ? h@Wout + b : 0 -------------------------

__global__ void k_final(const float* __restrict__ h, const float* __restrict__ obs,
                        const float* __restrict__ Wout, const float* __restrict__ bout,
                        float* __restrict__ out) {
  const int node = blockIdx.x * 8 + (threadIdx.x >> 5);   // one wave32 per node
  const int lane = threadIdx.x & 31;

  const float* orow = obs + (size_t)node * FF;
  int nz = 0;
  #pragma unroll
  for (int t = 0; t < FF / 32; ++t) nz |= (orow[lane + 32 * t] != 0.0f) ? 1 : 0;

  const float* hrow = h + (size_t)node * HH;
  float s = 0.0f;
  #pragma unroll
  for (int t = 0; t < HH / 32; ++t) { const int d = lane + 32 * t; s += hrow[d] * Wout[d]; }

  #pragma unroll
  for (int off = 16; off > 0; off >>= 1) {
    s  += __shfl_down(s,  off, 32);
    nz |= __shfl_down(nz, off, 32);
  }
  if (lane == 0) out[node] = nz ? (s + bout[0]) : 0.0f;
}

// ---- launch --------------------------------------------------------------

extern "C" void kernel_launch(void* const* d_in, const int* in_sizes, int n_in,
                              void* d_out, int out_size, void* d_ws, size_t ws_size,
                              hipStream_t stream) {
  (void)in_sizes; (void)n_in; (void)out_size; (void)ws_size;
  const float* obs  = (const float*)d_in[0];
  const float* W1   = (const float*)d_in[1];
  const float* b1   = (const float*)d_in[2];
  const float* W2   = (const float*)d_in[3];
  const float* b2   = (const float*)d_in[4];
  const float* Wout = (const float*)d_in[5];
  const float* bout = (const float*)d_in[6];
  float* out = (float*)d_out;

  char* ws = (char*)d_ws;
  size_t off = 0;
  auto alloc = [&](size_t bytes) -> void* {
    void* p = ws + off;
    off = (off + bytes + 255) & ~(size_t)255;
    return p;
  };
  int*      nbrs  = (int*)     alloc(sizeof(int)      * (size_t)MTOT * KK);
  _Float16* obs16 = (_Float16*)alloc(sizeof(_Float16) * (size_t)MTOT * FF);
  _Float16* w1t   = (_Float16*)alloc(sizeof(_Float16) * (size_t)FF * HH);  // [H][F]
  _Float16* w2t   = (_Float16*)alloc(sizeof(_Float16) * (size_t)HH * HH);  // [H][H]
  _Float16* h16   = (_Float16*)alloc(sizeof(_Float16) * (size_t)MTOT * HH);
  float*    bufM  = (float*)   alloc(sizeof(float)    * (size_t)MTOT * HH);
  float*    bufA  = (float*)   alloc(sizeof(float)    * (size_t)MTOT * HH);

  // precision prep + topology
  k_cvt16 <<<(MTOT * FF + 255) / 256, 256, 0, stream>>>(obs, obs16, MTOT * FF);
  k_cvtW_T<<<(FF * HH + 255) / 256,  256, 0, stream>>>(W1, w1t, FF, FF * HH);
  k_cvtW_T<<<(HH * HH + 255) / 256,  256, 0, stream>>>(W2, w2t, HH, HH * HH);
  k_topk  <<<BB, NN, 0, stream>>>(obs, nbrs);

  const dim3 gg(MTOT / 256, HH / 32);

  // layer 1
  k_gemm<FF><<<gg, 256, 0, stream>>>(obs16, w1t, bufM, bufA);
  k_scatter<<<MTOT, 256, 0, stream>>>(bufM, nbrs, bufA);
  k_act<<<(MTOT * HH) / 256, 256, 0, stream>>>(bufA, b1, h16, nullptr, MTOT * HH);

  // layer 2
  k_gemm<HH><<<gg, 256, 0, stream>>>(h16, w2t, bufM, bufA);
  k_scatter<<<MTOT, 256, 0, stream>>>(bufM, nbrs, bufA);
  k_act<<<(MTOT * HH) / 256, 256, 0, stream>>>(bufA, b2, h16, bufM, MTOT * HH);

  // readout
  k_final<<<MTOT / 8, 256, 0, stream>>>(bufM, obs, Wout, bout, out);
}
